// SESTN_47991964566195
// MI455X (gfx1250) — matrix operand order repaired and checked by reading, not compiled, not run on hardware
//
#include <hip/hip_runtime.h>

// ---------------------------------------------------------------------------
// Model dims
#define S_DIM   3
#define C_DIM   62
#define F_DIM   5
#define E_DIM   5
#define D_MODEL 310
#define D_INNER 620
#define D_STATE 16
#define DT_RANK 20
#define VOCABD  128
#define GDIM    65
#define BATCH   1024
#define HC      19          // E + N_HID + N_CLS_HG
#define HO      57          // S * HC

typedef __attribute__((ext_vector_type(16))) _Float16 v16h;
typedef __attribute__((ext_vector_type(8)))  _Float16 h8;
typedef __attribute__((ext_vector_type(8)))  float    v8f;

// ---------------------------------------------------------------------------
// Generic WMMA GEMM: C[M,N] (+=) act(A[M,K] @ B[K,N] + bias)
// 4 waves/block; 64x64 macro tile; each wave owns a 32x32 output tile =
// 4 accumulators -> 4 v_wmma_f32_16x16x32_f16 per K-chunk of 32.
// A staged row-major in LDS, B staged *transposed* (N-major) so both A and B
// fragments are two aligned 16-byte ds_load_b128 per 16x16 operand.
#define GTM 64
#define GTN 64
#define GTK 32
#define LSA 40              // halves per As row   (32 K + pad)
#define LSB 40              // halves per Bt row   (32 K + pad)

__launch_bounds__(128)
__global__ void gemm_f16_wmma(
    const float* __restrict__ Ag, long lda,
    const float* __restrict__ Bg, long ldb,
    float* __restrict__ Cg, long ldc,
    const float* __restrict__ bias,
    int M, int N, int K,
    int act, int addC,
    int n1, int n2, int s0, int s1, int s2, int rbase,
    int aMod, long aStr, long bStr, long cStr)
{
    const int z = blockIdx.z;
    Ag += (long)(aMod ? (z % aMod) : z) * aStr;
    Bg += (long)z * bStr;
    Cg += (long)z * cStr;

    __shared__ _Float16 As[GTM * LSA];   // [row(M)][k]
    __shared__ _Float16 Bt[GTN * LSB];   // [col(N)][k]  (transposed)

    const int tid  = threadIdx.x;
    const int lane = tid & 31;
    const int wave = tid >> 5;
    const int lrow = lane & 15;
    const int half = lane >> 4;
    const int m0 = blockIdx.x * GTM;
    const int n0 = blockIdx.y * GTN;
    const int wm = (wave >> 1) * 32;
    const int wn = (wave & 1) * 32;

    auto remap = [&](int m) -> long {
        int p  = n1 * n2;
        int i0 = m / p;
        int r_ = m - i0 * p;
        int i1 = r_ / n2;
        int i2 = r_ - i1 * n2;
        return (long)i0 * s0 + (long)i1 * s1 + (long)i2 * s2 + rbase;
    };

    v8f acc00 = {}, acc01 = {}, acc10 = {}, acc11 = {};
    const int gn0 = n0 + wn + lrow;
    const int gn1 = gn0 + 16;
    if (addC) {
        #pragma unroll
        for (int r = 0; r < 8; ++r) {
            int gmA = m0 + wm + r + 8 * half;
            int gmB = gmA + 16;
            if (gmA < M && gn0 < N) acc00[r] = Cg[remap(gmA) * ldc + gn0];
            if (gmA < M && gn1 < N) acc01[r] = Cg[remap(gmA) * ldc + gn1];
            if (gmB < M && gn0 < N) acc10[r] = Cg[remap(gmB) * ldc + gn0];
            if (gmB < M && gn1 < N) acc11[r] = Cg[remap(gmB) * ldc + gn1];
        }
    }

    const bool fullMN = (m0 + GTM <= M) && (n0 + GTN <= N);

    for (int k0 = 0; k0 < K; k0 += GTK) {
        if (fullMN && (k0 + GTK <= K)) {
            // ---------- fast path: unconditional, batched loads ----------
            #pragma unroll
            for (int p = 0; p < 2; ++p) {
                int slot = tid + p * 128;          // 0..255
                int r  = slot >> 2;                // 0..63  (M row)
                int c0 = (slot & 3) * 8;           // 0,8,16,24 (K offset)
                const float* ap = Ag + (long)(m0 + r) * lda + (k0 + c0);
                h8 hv;
                #pragma unroll
                for (int j = 0; j < 8; ++j) hv[j] = (_Float16)ap[j];
                *(h8*)(As + r * LSA + c0) = hv;
            }
            #pragma unroll
            for (int p = 0; p < 2; ++p) {
                int slot = tid + p * 128;
                int r  = slot >> 3;                // 0..31 (K row)
                int c0 = (slot & 7) * 8;           // N offset
                const float* bp = Bg + (long)(k0 + r) * ldb + (n0 + c0);
                float v[8];
                #pragma unroll
                for (int j = 0; j < 8; ++j) v[j] = bp[j];
                #pragma unroll
                for (int j = 0; j < 8; ++j) Bt[(c0 + j) * LSB + r] = (_Float16)v[j];
            }
        } else {
            // ---------- edge path: clamped address + select (no branches) ----
            for (int slot = tid; slot < GTM * GTK; slot += 128) {
                int r = slot >> 5, c = slot & 31;
                int gm = m0 + r, gk = k0 + c;
                bool ok = (gm < M) && (gk < K);
                float v = Ag[ok ? ((long)gm * lda + gk) : 0];
                As[r * LSA + c] = (_Float16)(ok ? v : 0.0f);
            }
            for (int slot = tid; slot < GTK * GTN; slot += 128) {
                int r = slot >> 6, c = slot & 63;
                int gk = k0 + r, gc = n0 + c;
                bool ok = (gk < K) && (gc < N);
                float v = Bg[ok ? ((long)gk * ldb + gc) : 0];
                Bt[c * LSB + r] = (_Float16)(ok ? v : 0.0f);
            }
        }
        if (k0 + GTK < K) {      // gfx1250 global_prefetch_b8 of next tiles
            __builtin_prefetch(Ag + (long)m0 * lda + k0 + GTK, 0, 0);
            __builtin_prefetch(Bg + (long)(k0 + GTK) * ldb + n0, 0, 0);
        }
        __syncthreads();

        // fragments: two aligned 16B LDS loads per operand (ISA layouts)
        const _Float16* ar0 = As + (wm + lrow) * LSA;
        const _Float16* ar1 = As + (wm + 16 + lrow) * LSA;
        const _Float16* bc0 = Bt + (wn + lrow) * LSB;
        const _Float16* bc1 = Bt + (wn + 16 + lrow) * LSB;

        h8 a0l = *(const h8*)(ar0 + 8 * half);
        h8 a0h = *(const h8*)(ar0 + 16 + 8 * half);
        h8 a1l = *(const h8*)(ar1 + 8 * half);
        h8 a1h = *(const h8*)(ar1 + 16 + 8 * half);
        h8 b0l = *(const h8*)(bc0 + 8 * half);
        h8 b0h = *(const h8*)(bc0 + 16 + 8 * half);
        h8 b1l = *(const h8*)(bc1 + 8 * half);
        h8 b1h = *(const h8*)(bc1 + 16 + 8 * half);

        v16h af0 = __builtin_shufflevector(a0l, a0h, 0,1,2,3,4,5,6,7,8,9,10,11,12,13,14,15);
        v16h af1 = __builtin_shufflevector(a1l, a1h, 0,1,2,3,4,5,6,7,8,9,10,11,12,13,14,15);
        v16h bf0 = __builtin_shufflevector(b0l, b0h, 0,1,2,3,4,5,6,7,8,9,10,11,12,13,14,15);
        v16h bf1 = __builtin_shufflevector(b1l, b1h, 0,1,2,3,4,5,6,7,8,9,10,11,12,13,14,15);

        acc00 = __builtin_amdgcn_wmma_f32_16x16x32_f16(false, af0, false, bf0, (short)0, acc00, false, false);
        acc01 = __builtin_amdgcn_wmma_f32_16x16x32_f16(false, af0, false, bf1, (short)0, acc01, false, false);
        acc10 = __builtin_amdgcn_wmma_f32_16x16x32_f16(false, af1, false, bf0, (short)0, acc10, false, false);
        acc11 = __builtin_amdgcn_wmma_f32_16x16x32_f16(false, af1, false, bf1, (short)0, acc11, false, false);
        __syncthreads();
    }

    auto epi = [&](float v, int gc) -> float {
        if (bias) v += bias[gc];
        if (act == 1)      v = fmaxf(v, 0.0f);
        else if (act == 2) v = v / (1.0f + __expf(-v));
        else if (act == 3) v = fmaxf(v, 0.0f) + log1pf(__expf(-fabsf(v)));
        return v;
    };
    #pragma unroll
    for (int r = 0; r < 8; ++r) {
        int gmA = m0 + wm + r + 8 * half;
        int gmB = gmA + 16;
        if (gmA < M && gn0 < N) Cg[remap(gmA) * ldc + gn0] = epi(acc00[r], gn0);
        if (gmA < M && gn1 < N) Cg[remap(gmA) * ldc + gn1] = epi(acc01[r], gn1);
        if (gmB < M && gn0 < N) Cg[remap(gmB) * ldc + gn0] = epi(acc10[r], gn0);
        if (gmB < M && gn1 < N) Cg[remap(gmB) * ldc + gn1] = epi(acc11[r], gn1);
    }
}

// ---------------------------------------------------------------------------
// Fused hypergraph + HGNN: one block per (b,s,f); sim/G live in LDS only.
__launch_bounds__(256)
__global__ void hyper_kernel(const float* __restrict__ x,
                             const float* __restrict__ ce,
                             const float* __restrict__ EAdj,
                             const float* __restrict__ W1, const float* __restrict__ b1,
                             const float* __restrict__ W2, const float* __restrict__ b2,
                             float* __restrict__ gOut, float* __restrict__ fx,
                             float* __restrict__ hout)
{
    const int blk = blockIdx.x;
    const int f = blk % F_DIM;
    const int s = (blk / F_DIM) % S_DIM;
    const int b = blk / (F_DIM * S_DIM);

    __shared__ float simL[C_DIM * C_DIM];
    __shared__ float GL[C_DIM * C_DIM];
    __shared__ float x5L[C_DIM * E_DIM];
    __shared__ float anL[C_DIM * E_DIM];
    __shared__ float wL[C_DIM], DeL[C_DIM], dviL[C_DIM];
    __shared__ float t1L[C_DIM * 6], h1L[C_DIM * 6];
    __shared__ float t2L[C_DIM * 8], h2L[C_DIM * 8];

    const int tid = threadIdx.x;

    if (tid < C_DIM) {
        float xv = x[(((long)b * S_DIM + s) * C_DIM + tid) * F_DIM + f];
        float v[E_DIM]; float nrm = 0.f;
        #pragma unroll
        for (int e = 0; e < E_DIM; ++e) {
            v[e] = xv * ce[(s * F_DIM + f) * E_DIM + e];
            nrm += v[e] * v[e];
        }
        nrm = fmaxf(sqrtf(nrm), 1e-8f);
        #pragma unroll
        for (int e = 0; e < E_DIM; ++e) { x5L[tid*E_DIM+e] = v[e]; anL[tid*E_DIM+e] = v[e]/nrm; }
        wL[tid] = EAdj[f * (S_DIM * C_DIM) + s * C_DIM + tid];
    }
    __syncthreads();

    for (int idx = tid; idx < C_DIM * C_DIM; idx += 256) {
        int c = idx / C_DIM, d = idx % C_DIM;
        float dot = 0.f;
        #pragma unroll
        for (int e = 0; e < E_DIM; ++e) dot += anL[c*E_DIM+e] * anL[d*E_DIM+e];
        simL[idx] = (dot + 1.0f) * 0.5f;
    }
    __syncthreads();

    if (tid < C_DIM) {
        float sd = 0.f, sv = 0.f;
        for (int c = 0; c < C_DIM; ++c) sd += simL[c * C_DIM + tid];         // De[d]
        DeL[tid] = fmaxf(sd, 1e-8f);
        for (int d = 0; d < C_DIM; ++d) sv += simL[tid * C_DIM + d] * wL[d]; // Dv[c]
        dviL[tid] = rsqrtf(fmaxf(sv, 1e-8f));
    }
    __syncthreads();

    const long gbase = ((((long)f * BATCH + b) * S_DIM + s) * C_DIM) * C_DIM;
    for (int idx = tid; idx < C_DIM * C_DIM; idx += 256) {
        int c = idx / C_DIM, d = idx % C_DIM;
        float a = 0.f;
        for (int e = 0; e < C_DIM; ++e)
            a += (simL[c*C_DIM+e] * wL[e] / DeL[e]) * simL[d*C_DIM+e];
        float g = dviL[c] * a * dviL[d];
        GL[idx] = g;
        gOut[gbase + idx] = g;
    }
    __syncthreads();

    for (int idx = tid; idx < C_DIM * 6; idx += 256) {
        int c = idx / 6, j = idx % 6;
        float a = b1[j];
        #pragma unroll
        for (int e = 0; e < E_DIM; ++e) a += x5L[c*E_DIM+e] * W1[e*6+j];
        t1L[idx] = a;
    }
    __syncthreads();
    for (int idx = tid; idx < C_DIM * 6; idx += 256) {
        int c = idx / 6, j = idx % 6;
        float a = 0.f;
        for (int d = 0; d < C_DIM; ++d) a += GL[c*C_DIM+d] * t1L[d*6+j];
        h1L[idx] = fmaxf(a, 0.f);
    }
    __syncthreads();
    for (int idx = tid; idx < C_DIM * 8; idx += 256) {
        int c = idx / 8, j = idx % 8;
        float a = b2[j];
        #pragma unroll
        for (int k = 0; k < 6; ++k) a += h1L[c*6+k] * W2[k*8+j];
        t2L[idx] = a;
    }
    __syncthreads();
    for (int idx = tid; idx < C_DIM * 8; idx += 256) {
        int c = idx / 8, j = idx % 8;
        float a = 0.f;
        for (int d = 0; d < C_DIM; ++d) a += GL[c*C_DIM+d] * t2L[d*8+j];
        h2L[idx] = a;
    }
    __syncthreads();

    const long fb = (((long)f * BATCH + b) * S_DIM + s) * D_MODEL;
    for (int idx = tid; idx < C_DIM * E_DIM; idx += 256) fx[fb + idx] = x5L[idx];
    for (int c = tid; c < C_DIM; c += 256) {
        long hb = (((long)f * BATCH + b) * C_DIM + c) * HO + s * HC;
        #pragma unroll
        for (int e = 0; e < E_DIM; ++e) hout[hb + e] = x5L[c*E_DIM+e];
        #pragma unroll
        for (int j = 0; j < 6; ++j) hout[hb + 5 + j] = h1L[c*6+j];
        #pragma unroll
        for (int j = 0; j < 8; ++j) hout[hb + 11 + j] = h2L[c*8+j];
    }
}

// ---------------------------------------------------------------------------
__global__ void rmsnorm_kernel(const float* __restrict__ in, const float* __restrict__ w,
                               float* __restrict__ out, int R, int D)
{
    __shared__ float red[256];
    int row = blockIdx.x;
    if (row >= R) return;
    const float* xr = in + (long)row * D;
    float s = 0.f;
    for (int i = threadIdx.x; i < D; i += 256) { float v = xr[i]; s += v * v; }
    red[threadIdx.x] = s; __syncthreads();
    for (int o = 128; o > 0; o >>= 1) {
        if (threadIdx.x < o) red[threadIdx.x] += red[threadIdx.x + o];
        __syncthreads();
    }
    float rs = rsqrtf(red[0] / D + 1e-5f);
    for (int i = threadIdx.x; i < D; i += 256) out[(long)row * D + i] = xr[i] * rs * w[i];
}

__global__ void conv_silu_kernel(const float* __restrict__ xz, const float* __restrict__ cw,
                                 const float* __restrict__ cb, float* __restrict__ xi)
{
    int idx = blockIdx.x * blockDim.x + threadIdx.x;
    if (idx >= BATCH * D_INNER) return;
    int b = idx / D_INNER, d = idx % D_INNER;
    const float* base = xz + (long)b * S_DIM * (2 * D_INNER) + d;
    float x0 = base[0], x1 = base[2*D_INNER], x2 = base[4*D_INNER];
    float w1 = cw[d*4+1], w2 = cw[d*4+2], w3 = cw[d*4+3];
    float bb = cb[d];
    float c0 = x0*w3 + bb;
    float c1 = x0*w2 + x1*w3 + bb;
    float c2 = x0*w1 + x1*w2 + x2*w3 + bb;
    long o = (long)b * S_DIM * D_INNER + d;
    xi[o]             = c0 / (1.f + __expf(-c0));
    xi[o +   D_INNER] = c1 / (1.f + __expf(-c1));
    xi[o + 2*D_INNER] = c2 / (1.f + __expf(-c2));
}

__global__ void ssm_scan_kernel(const float* __restrict__ dt, const float* __restrict__ proj,
                                const float* __restrict__ xi, const float* __restrict__ xz,
                                const float* __restrict__ A_log, const float* __restrict__ Dp,
                                float* __restrict__ y)
{
    int idx = blockIdx.x * blockDim.x + threadIdx.x;
    if (idx >= BATCH * D_INNER) return;
    int b = idx / D_INNER, d = idx % D_INNER;
    float Ad[D_STATE], h[D_STATE];
    #pragma unroll
    for (int n = 0; n < D_STATE; ++n) { Ad[n] = -__expf(A_log[d*D_STATE+n]); h[n] = 0.f; }
    float Dd = Dp[d];
    #pragma unroll
    for (int s = 0; s < S_DIM; ++s) {
        long r = (long)b * S_DIM + s;
        float dtv = dt[r * D_INNER + d];
        float xiv = xi[r * D_INNER + d];
        const float* pr = proj + r * (DT_RANK + 2 * D_STATE);
        float acc = 0.f;
        #pragma unroll
        for (int n = 0; n < D_STATE; ++n) {
            float dA = __expf(dtv * Ad[n]);
            h[n] = dA * h[n] + dtv * pr[DT_RANK + n] * xiv;
            acc += h[n] * pr[DT_RANK + D_STATE + n];
        }
        acc += Dd * xiv;
        float zv = xz[r * (2 * D_INNER) + D_INNER + d];
        float sz = zv / (1.f + __expf(-zv));
        y[r * D_INNER + d] = acc * sz;
    }
}

__global__ void bn_kernel(float* __restrict__ t, const float* __restrict__ g,
                          const float* __restrict__ b, int f, int n)
{
    int i = blockIdx.x * blockDim.x + threadIdx.x;
    if (i < n) {
        float sc = g[f] * rsqrtf(1.0f + 1e-5f);
        t[i] = t[i] * sc + b[f];
    }
}

__global__ void softmax14_kernel(const float* __restrict__ in, float* __restrict__ out, int R)
{
    int r = blockIdx.x * blockDim.x + threadIdx.x;
    if (r >= R) return;
    const float* xr = in + (long)r * 14;
    float m = xr[0];
    #pragma unroll
    for (int i = 1; i < 14; ++i) m = fmaxf(m, xr[i]);
    float e[14], s = 0.f;
    #pragma unroll
    for (int i = 0; i < 14; ++i) { e[i] = __expf(xr[i] - m); s += e[i]; }
    float inv = 1.f / s;
    #pragma unroll
    for (int i = 0; i < 14; ++i) out[(long)r * 14 + i] = e[i] * inv;
}

// ---------------------------------------------------------------------------
static inline void gemm(hipStream_t st, const float* A, long lda, const float* B, long ldb,
                        float* C, long ldc, const float* bias, int M, int N, int K,
                        int act = 0, int addC = 0,
                        int n1 = 1, int n2 = 1, int s0 = 1, int s1 = 0, int s2 = 0,
                        int rbase = 0, int batch = 1, int aMod = 0,
                        long aStr = 0, long bStr = 0, long cStr = 0)
{
    dim3 g((M + GTM - 1) / GTM, (N + GTN - 1) / GTN, batch);
    gemm_f16_wmma<<<g, dim3(128), 0, st>>>(A, lda, B, ldb, C, ldc, bias, M, N, K,
                                           act, addC, n1, n2, s0, s1, s2, rbase,
                                           aMod, aStr, bStr, cStr);
}

extern "C" void kernel_launch(void* const* d_in, const int* in_sizes, int n_in,
                              void* d_out, int out_size, void* d_ws, size_t ws_size,
                              hipStream_t stream)
{
    (void)in_sizes; (void)n_in; (void)out_size;
#define P(i) ((const float*)d_in[(i)])
    const float* x       = P(0);
    const float* EAdj    = P(1);
    const float* STAdj   = P(2);
    const float* bn_b    = P(3);
    const float* bn_g    = P(4);
    const float* cont_emb= P(5);
    const float* dom_W   = P(6);
    const float* dom_b   = P(7);
    const float* fc_W    = P(8);
    const float* fc_b    = P(9);
    const float* fq_W1   = P(10);
    const float* fq_W2   = P(11);
    const float* fq_b1   = P(12);
    const float* fq_b2   = P(13);
    const float* gnn_W1  = P(14);
    const float* gnn_W2  = P(15);
    const float* gnn_b1  = P(16);
    const float* gnn_b2  = P(17);
    const float* hg_W1   = P(18);
    const float* hg_W2   = P(19);
    const float* hg_b1   = P(20);
    const float* hg_b2   = P(21);
    const float* spa_W   = P(132);
    const float* spa_b   = P(133);
    const float* tem_W   = P(134);
    const float* tem_b   = P(135);

    // ---- workspace layout (floats) ----
    const long FX_OFF   = 0;                                               // (F,B,S,310)
    const long HOUT_OFF = FX_OFF   + (long)F_DIM*BATCH*S_DIM*D_MODEL;      // 4,761,600
    const long CAT_OFF  = HOUT_OFF + (long)F_DIM*BATCH*C_DIM*HO;           // +18,094,080
    const long G1_OFF   = CAT_OFF  + (long)BATCH*F_DIM*GDIM*256;           // +85,196,800
    const long T2_OFF   = G1_OFF   + (long)BATCH*F_DIM*GDIM*64;            // +21,299,200
    const long AUX_OFF  = T2_OFF   + (long)BATCH*F_DIM*GDIM*128;           // +42,598,400
    const long CO_OFF   = AUX_OFF  + 21299200L;
    const long DL_OFF   = CO_OFF   + (long)BATCH*128;
    const long WS_END   = DL_OFF   + (long)BATCH*14;
    if (ws_size < (size_t)WS_END * sizeof(float)) return;

    float* ws   = (float*)d_ws;
    float* fx   = ws + FX_OFF;
    float* hout = ws + HOUT_OFF;
    float* cat  = ws + CAT_OFF;
    float* g1b  = ws + G1_OFF;
    float* t2b  = ws + T2_OFF;
    float* aux  = ws + AUX_OFF;
    float* cout = ws + CO_OFF;
    float* dlog = ws + DL_OFF;

    // mamba scratch inside aux
    float* HB  = aux;
    float* XN  = aux + 952320L;
    float* XZ  = aux + 1904640L;
    float* XI  = aux + 5713920L;
    float* PRJ = aux + 7618560L;
    float* DTb = aux + 7778304L;
    float* YB  = aux + 9682944L;
    float* TO  = aux + 11587584L;
    float* T1b = aux;                 // reused after mamba
    float* SQ  = aux;                 // reused after g1

    // ---- output layout (floats) ----
    float* dout = (float*)d_out;
    const long OUT_OFF   = 0;
    const long DOM_OFF   = 3072;
    const long STADJ_OFF = 17408;
    const long G_OFF     = 38533;
    const long FRE_OFF   = G_OFF + (long)F_DIM*BATCH*S_DIM*C_DIM*C_DIM;   // 59,082,373

    // ========== Stage 1: fused hypergraph + HGNN (writes G, fx, hout) ======
    hyper_kernel<<<BATCH * S_DIM * F_DIM, 256, 0, stream>>>(
        x, cont_emb, EAdj, hg_W1, hg_b1, hg_W2, hg_b2,
        dout + G_OFF, fx, hout);

    // ========== Stage 2: spatial projection directly into cat =============
    // m = (f*B + b)*C + c  ->  cat row b*(F*65) + f*65 + c
    gemm(stream, hout, HO, spa_W, 256, cat, 256, spa_b,
         F_DIM * BATCH * C_DIM, 256, HO, 0, 0,
         BATCH, C_DIM, /*s0=*/GDIM, /*s1=*/F_DIM * GDIM, /*s2=*/1, 0);

    // ========== Stage 3: mamba stacks, f = 0..4 ===========================
    for (int f = 0; f < F_DIM; ++f) {
        int mb = 22 + f * 22;
        const float* fxf = fx + (long)f * BATCH * S_DIM * D_MODEL;
        hipMemcpyAsync(HB, fxf, (size_t)BATCH * S_DIM * D_MODEL * sizeof(float),
                       hipMemcpyDeviceToDevice, stream);
        for (int l = 0; l < 2; ++l) {
            int lb = mb + l * 10;
            const float* A_log = P(lb+0); const float* Dp    = P(lb+1);
            const float* Wdt   = P(lb+2); const float* Win   = P(lb+3);
            const float* Wout  = P(lb+4); const float* Wx    = P(lb+5);
            const float* bdt   = P(lb+6); const float* convb = P(lb+7);
            const float* convw = P(lb+8); const float* norm  = P(lb+9);

            rmsnorm_kernel<<<BATCH * S_DIM, 256, 0, stream>>>(HB, norm, XN, BATCH*S_DIM, D_MODEL);
            gemm(stream, XN, D_MODEL, Win, 2*D_INNER, XZ, 2*D_INNER, nullptr,
                 BATCH*S_DIM, 2*D_INNER, D_MODEL);
            conv_silu_kernel<<<(BATCH*D_INNER + 255)/256, 256, 0, stream>>>(XZ, convw, convb, XI);
            gemm(stream, XI, D_INNER, Wx, DT_RANK + 2*D_STATE, PRJ, DT_RANK + 2*D_STATE,
                 nullptr, BATCH*S_DIM, DT_RANK + 2*D_STATE, D_INNER);
            gemm(stream, PRJ, DT_RANK + 2*D_STATE, Wdt, D_INNER, DTb, D_INNER, bdt,
                 BATCH*S_DIM, D_INNER, DT_RANK, /*act=softplus*/3);
            ssm_scan_kernel<<<(BATCH*D_INNER + 255)/256, 256, 0, stream>>>(
                DTb, PRJ, XI, XZ, A_log, Dp, YB);
            gemm(stream, YB, D_INNER, Wout, D_MODEL, HB, D_MODEL, nullptr,
                 BATCH*S_DIM, D_MODEL, D_INNER, 0, /*addC=*/1);
        }
        const float* lm_W   = P(mb + 20);
        const float* norm_f = P(mb + 21);
        rmsnorm_kernel<<<BATCH * S_DIM, 256, 0, stream>>>(HB, norm_f, XN, BATCH*S_DIM, D_MODEL);
        gemm(stream, XN, D_MODEL, lm_W, VOCABD, TO, VOCABD, nullptr, BATCH*S_DIM, VOCABD, D_MODEL);
        bn_kernel<<<(BATCH*S_DIM*VOCABD + 255)/256, 256, 0, stream>>>(
            TO, bn_g, bn_b, f, BATCH*S_DIM*VOCABD);
        // m = b*S + s  ->  cat row b*(F*65) + f*65 + 62 + s
        gemm(stream, TO, VOCABD, tem_W, 256, cat, 256, tem_b,
             BATCH*S_DIM, 256, VOCABD, 0, 0,
             1, S_DIM, /*s0=*/F_DIM*GDIM, 0, /*s2=*/1, /*rbase=*/f*GDIM + C_DIM);
    }

    // ========== Stage 4: GNN over cat (B,F,65,256) ========================
    const int MB = BATCH * F_DIM * GDIM;          // 332800
    gemm(stream, cat, 256, gnn_W1, 64, T1b, 64, gnn_b1, MB, 64, 256);
    // g1 = relu(STAdj[f] @ t1[b,f])   batch z = b*F + f
    gemm(stream, STAdj, GDIM, T1b, 64, g1b, 64, nullptr, GDIM, 64, GDIM,
         /*act=*/1, 0, 1, 1, 1, 0, 0, 0,
         BATCH*F_DIM, /*aMod=*/F_DIM, (long)GDIM*GDIM, (long)GDIM*64, (long)GDIM*64);
    gemm(stream, g1b, 64, gnn_W2, 128, t2b, 128, gnn_b2, MB, 128, 64);
    // fre = STAdj[f] @ t2 -> d_out
    gemm(stream, STAdj, GDIM, t2b, 128, dout + FRE_OFF, 128, nullptr, GDIM, 128, GDIM,
         0, 0, 1, 1, 1, 0, 0, 0,
         BATCH*F_DIM, F_DIM, (long)GDIM*GDIM, (long)GDIM*128, (long)GDIM*128);

    // ========== Stage 5: heads ============================================
    gemm(stream, dout + FRE_OFF, 128, fq_W1, 40, SQ, 40, fq_b1, MB, 40, 128);
    gemm(stream, SQ, (long)F_DIM*GDIM*40, fq_W2, 128, cout, 128, fq_b2,
         BATCH, 128, F_DIM*GDIM*40);
    gemm(stream, cout, 128, fc_W, 3, dout + OUT_OFF, 3, fc_b, BATCH, 3, 128);
    gemm(stream, cout, 128, dom_W, 14, dlog, 14, dom_b, BATCH, 14, 128);
    softmax14_kernel<<<(BATCH + 255)/256, 256, 0, stream>>>(dlog, dout + DOM_OFF, BATCH);
    hipMemcpyAsync(dout + STADJ_OFF, STAdj, (size_t)F_DIM*GDIM*GDIM*sizeof(float),
                   hipMemcpyDeviceToDevice, stream);
#undef P
}
